// ROMAN_R_Layer_72138270703831
// MI455X (gfx1250) — compile-verified, matrix operand order checked by reading, not run on hardware
//
#include <hip/hip_runtime.h>
#include <hip/hip_bf16.h>

// ---------------------------------------------------------------------------
// MI455X (gfx1250) ROMAN-R layer.
// All six 3D convs + depthwise conv run through v_wmma_f32_16x16x32_bf16.
// 1->32 convs: im2col GEMM with K' = (dt,dh)-row*8 + dw (15 rows, taps 5..7
//   zero-weighted): A build = uniform row base + contiguous immediate loads.
// 32->1 convs + depthwise: Toeplitz-along-W GEMM; 32->1 A operand = 8 dword loads
//   bit-cast straight into the operand registers (padded rows, stride 136).
// All B operands come from LDS tables of pre-packed 32-bit bf16 PAIRS with
// hardwired zero slots: 8 unconditional ds_load_b32 per WMMA, no half-packing.
// SVT uses the Gram-matrix trick (16x16 Jacobi eigensolve) since T=16 << HW.
// ---------------------------------------------------------------------------

#define Bn 4
#define Tn 16
#define Hn 128
#define Wn 128
#define HIDn 32
#define HWn (Hn * Wn)
#define NPIX (Bn * Tn * Hn * Wn)          // 1048576
#define NCPIX (Bn * HIDn * Tn * Hn * Wn)  // 33554432

// Padded bf16 row layout: stride 136 (4-element replicate halo each side)
#define RSp 136
#define PLp (Hn * RSp)   // 17408 (one frame)
#define CSp (Tn * PLp)   // 278528 (one channel)
#define PAD_HALFS ((size_t)Bn * HIDn * CSp)       // 32-channel padded tensor
#define SPAD_HALFS ((size_t)Bn * Tn * Hn * RSp)   // 1-channel padded tensor
#define SLACK_HALFS 128
#define PADBUF_HALFS (PAD_HALFS + SLACK_HALFS)
#define SPADBUF_HALFS (SPAD_HALFS + SLACK_HALFS)

typedef __attribute__((ext_vector_type(16))) __bf16 v16bf;
typedef __attribute__((ext_vector_type(8))) unsigned int u32x8;
typedef __attribute__((ext_vector_type(8))) float v8f;

__device__ __forceinline__ int iclamp(int x, int lo, int hi) {
  return x < lo ? lo : (x > hi ? hi : x);
}

__device__ __forceinline__ unsigned short f2bf(float f) {
  __hip_bfloat16 h = __float2bfloat16(f);
  return __builtin_bit_cast(unsigned short, h);
}

__device__ __forceinline__ float bf2f(unsigned short u) {
  union { unsigned u; float f; } c;
  c.u = ((unsigned)u) << 16;
  return c.f;
}

__device__ __forceinline__ unsigned pack2(unsigned short lo, unsigned short hi) {
  return (unsigned)lo | ((unsigned)hi << 16);
}

__device__ __forceinline__ v8f wmma_bf16(v16bf a, v16bf b, v8f c) {
  return __builtin_amdgcn_wmma_f32_16x16x32_bf16(false, a, false, b, (short)0, c,
                                                 false, false);
}

// Operand pair index for VGPR i of a 16-bit wave32 operand:
// lanes 0-15 hold K pairs {0..3, 8..11}; lanes 16-31 hold {4..7, 12..15}.
__device__ __forceinline__ int pidx(int lane, int i) {
  int lop = (lane & 16) ? 4 : 0;
  return lop + ((i < 4) ? i : (i + 4));
}

__device__ __forceinline__ float softplusf(float x) {
  return (x > 20.f) ? x : log1pf(expf(x));
}

// Wave-uniform padded row offsets for the 15 (dt,dh) im2col rows of a
// 1-channel frame-major padded tensor; slot 15 redirects to a safe row.
__device__ __forceinline__ void rowbase15(int b, int t, int h, int* rowbase) {
#pragma unroll
  for (int r = 0; r < 16; ++r) {
    int rr = (r < 15) ? r : 14;
    int dt = rr / 5, dh = rr - dt * 5;
    int tt = iclamp(t + dt - 1, 0, Tn - 1);
    int hh = iclamp(h + dh - 2, 0, Hn - 1);
    rowbase[r] = ((b * Tn + tt) * Hn + hh) * RSp;
  }
}

// ---------------------------------------------------------------------------
__global__ void k_slack(unsigned short* __restrict__ a, unsigned short* __restrict__ b,
                        unsigned short* __restrict__ c, unsigned short* __restrict__ d,
                        unsigned short* __restrict__ e) {
  int tid = threadIdx.x;
  if (tid < SLACK_HALFS) {
    a[PAD_HALFS + tid] = 0;
    b[PAD_HALFS + tid] = 0;
    c[SPAD_HALFS + tid] = 0;
    d[SPAD_HALFS + tid] = 0;
    e[SPAD_HALFS + tid] = 0;
  }
}

// ---------------------------------------------------------------------------
// f32 -> padded bf16 (replicate halos) for the input M
// ---------------------------------------------------------------------------
__global__ void k_tobf16_pad(const float* __restrict__ src,
                             unsigned short* __restrict__ dst) {
  int n = blockIdx.x * 256 + threadIdx.x;
  int w = n & 127, rest = n >> 7;
  unsigned short v = f2bf(src[n]);
  size_t pi = (size_t)rest * RSp + 4 + w;
  dst[pi] = v;
  if (w == 0) {
#pragma unroll
    for (int q = 0; q < 4; ++q) dst[pi - 4 + q] = v;
  }
  if (w == 127) {
#pragma unroll
    for (int q = 0; q < 4; ++q) dst[pi + 1 + q] = v;
  }
}

// Helper: K' weight value (row*8+dw layout) for a (32,1,3,5,5) tensor
__device__ __forceinline__ unsigned short wkp(const float* __restrict__ W, int o, int k,
                                              float scale) {
  int row = k >> 3, dw = k & 7;
  if (row >= 15 || dw >= 5) return 0;
  int dt = row / 5, dh = row - dt * 5;
  return f2bf(scale * W[o * 75 + dt * 25 + dh * 5 + dw]);
}

// ---------------------------------------------------------------------------
// Mc = conv3d(M, W_convM)  (1 -> 32 ch, 3x5x5, replicate pad)
// im2col GEMM, K' = row*8+dw (4 chunks of 32). B table: pair-packed words.
// ---------------------------------------------------------------------------
__global__ void k_convM(const unsigned short* __restrict__ Mp,
                        const float* __restrict__ Wm, unsigned short* __restrict__ Mc) {
  __shared__ unsigned int Bw[64 * 32];  // [K-pair][out-channel]
  int tid = threadIdx.x;
  for (int idx = tid; idx < 64 * 32; idx += 256) {
    int p = idx >> 5, o = idx & 31;
    Bw[idx] = pack2(wkp(Wm, o, 2 * p, 1.f), wkp(Wm, o, 2 * p + 1, 1.f));
  }
  __syncthreads();

  int lane = tid & 31;
  int wt = blockIdx.x * 8 + (tid >> 5);
  int w0 = (wt & 7) << 4;
  int h = (wt >> 3) & 127;
  int t = (wt >> 10) & 15;
  int b = wt >> 14;
  int ar = lane & 15;
  int lo = (lane & 16) ? 8 : 0;
  int n0 = lane & 15;

  int rowbase[16];
  rowbase15(b, t, h, rowbase);
  int wcol = w0 + ar + 2;
  int pi8[8];
#pragma unroll
  for (int i = 0; i < 8; ++i) pi8[i] = pidx(lane, i);

  v8f acc0 = {}, acc1 = {};
#pragma unroll
  for (int kc = 0; kc < 4; ++kc) {
    int kb = kc * 32;
    int r1 = (kb + lo) >> 3;
    const unsigned short* p1 = Mp + rowbase[r1] + wcol;
    const unsigned short* p2 = Mp + rowbase[r1 + 2] + wcol;
    unsigned short au[16];
#pragma unroll
    for (int i = 0; i < 8; ++i) {
      au[i] = p1[i];
      au[8 + i] = p2[i];
    }
    u32x8 avv;
#pragma unroll
    for (int i = 0; i < 8; ++i) avv[i] = pack2(au[2 * i], au[2 * i + 1]);
    v16bf a = __builtin_bit_cast(v16bf, avv);
    u32x8 bv;
#pragma unroll
    for (int i = 0; i < 8; ++i) bv[i] = Bw[(kc * 16 + pi8[i]) * 32 + n0];
    acc0 = wmma_bf16(a, __builtin_bit_cast(v16bf, bv), acc0);
#pragma unroll
    for (int i = 0; i < 8; ++i) bv[i] = Bw[(kc * 16 + pi8[i]) * 32 + n0 + 16];
    acc1 = wmma_bf16(a, __builtin_bit_cast(v16bf, bv), acc1);
  }

  int mb = (lane >> 4) << 3;
  size_t base0 = (size_t)(b * HIDn + n0) * CSp + (size_t)t * PLp + (size_t)h * RSp + 4 + w0;
  size_t base1 = base0 + (size_t)16 * CSp;
#pragma unroll
  for (int jv = 0; jv < 8; ++jv) {
    Mc[base0 + mb + jv] = f2bf(acc0[jv]);
    Mc[base1 + mb + jv] = f2bf(acc1[jv]);
  }
  if (w0 == 0 && mb == 0) {
    unsigned short e0 = f2bf(acc0[0]), e1 = f2bf(acc1[0]);
#pragma unroll
    for (int q = 0; q < 4; ++q) { Mc[base0 - 4 + q] = e0; Mc[base1 - 4 + q] = e1; }
  }
  if (w0 == 112 && mb == 8) {
    unsigned short e0 = f2bf(acc0[7]), e1 = f2bf(acc1[7]);
#pragma unroll
    for (int q = 0; q < 4; ++q) { Mc[base0 + 16 + q] = e0; Mc[base1 + 16 + q] = e1; }
  }
}

// ---------------------------------------------------------------------------
// 32->1 conv core, Toeplitz GEMM over W on a padded bf16 tensor.
// Inner body: 2 pointer adds + 8 dword A loads + 8 ds_load_b32 pair-words + WMMA.
// Wp layout: [c][dt][dh][8 pair-words], pair[d'] = (w[d'-1], w[d']); 6,7 = 0.
// ---------------------------------------------------------------------------
__device__ v8f conv_c32_tile_bf(const unsigned short* __restrict__ In,
                                const unsigned int* __restrict__ Wp, int b, int t,
                                int h0, int w0, int lane) {
  int ar = lane & 15;  // A row (h offset)
  int j = lane & 15;   // B column (w offset)
  int lo = (lane & 16) ? 8 : 0;
  int psel[8];
#pragma unroll
  for (int i = 0; i < 8; ++i) {
    int d = 2 * pidx(lane, i) - j;             // tap of the pair's low half
    psel[i] = (d >= -1 && d <= 4) ? (d + 1) : 6;  // 6 -> zero word
  }
  int hrow[5];
#pragma unroll
  for (int dh = 0; dh < 5; ++dh) hrow[dh] = iclamp(h0 + ar + dh - 2, 0, Hn - 1) * RSp;
  int tofs[3];
#pragma unroll
  for (int dt = 0; dt < 3; ++dt) tofs[dt] = iclamp(t + dt - 1, 0, Tn - 1) * PLp;
  int wbase = w0 + lo + 2;  // even -> dword aligned

  v8f acc = {};
  size_t cb = (size_t)(b * HIDn) * CSp;
  for (int c = 0; c < HIDn; ++c, cb += CSp) {
    __builtin_prefetch(In + cb + CSp + tofs[0] + hrow[0] + wbase, 0, 0);
    const unsigned int* wc = Wp + c * 120;  // 15 rows * 8 words
    for (int dt = 0; dt < 3; ++dt) {
      const unsigned short* fbase = In + cb + tofs[dt] + wbase;
      for (int dh = 0; dh < 5; ++dh) {
        const unsigned int* p = (const unsigned int*)(fbase + hrow[dh]);
        u32x8 av;
#pragma unroll
        for (int i = 0; i < 4; ++i) {
          av[i] = p[i];
          av[4 + i] = p[8 + i];
        }
        const unsigned int* wrow = wc + (dt * 5 + dh) * 8;
        u32x8 bv;
#pragma unroll
        for (int i = 0; i < 8; ++i) bv[i] = wrow[psel[i]];
        acc = wmma_bf16(__builtin_bit_cast(v16bf, av), __builtin_bit_cast(v16bf, bv),
                        acc);
      }
    }
  }
  return acc;
}

// Stage a (32,1,3,5,5) weight tensor into the 8-pair-word LDS layout.
__device__ __forceinline__ void stage_wp5(unsigned int* Wp, const float* __restrict__ W,
                                          int tid) {
  for (int idx = tid; idx < HIDn * 120; idx += 256) {
    int c = idx / 120, r = idx - c * 120;
    int row = r >> 3, dp = r & 7;  // row = dt*5+dh, dp = pair word index
    const float* wr = W + c * 75 + (row / 5) * 25 + (row - (row / 5) * 5) * 5;
    unsigned short lo = (dp >= 1 && dp <= 5) ? f2bf(wr[dp - 1]) : (unsigned short)0;
    unsigned short hi = (dp < 5) ? f2bf(wr[dp]) : (unsigned short)0;
    Wp[idx] = pack2(lo, hi);
  }
}

// ---------------------------------------------------------------------------
// bg = 1 - (conv0(Mc)+b0); Ltmp = L - tl*bg^2*L + tl*bg^2*X - (tl/(rho+eps))*bg*U
// ---------------------------------------------------------------------------
__global__ void k_conv0_ltmp(const unsigned short* __restrict__ Mc,
                             const float* __restrict__ W0, const float* __restrict__ b0,
                             const float* __restrict__ X, const float* __restrict__ L,
                             const float* __restrict__ U,
                             const float* __restrict__ tau_l_p,
                             const float* __restrict__ rho_p, float* __restrict__ Ltmp) {
  __shared__ unsigned int Wp[HIDn * 120];
  int tid = threadIdx.x;
  stage_wp5(Wp, W0, tid);
  __syncthreads();

  int lane = tid & 31;
  int wt = blockIdx.x * 8 + (tid >> 5);
  int w0 = (wt & 7) << 4;
  int h0 = ((wt >> 3) & 7) << 4;
  int t = (wt >> 6) & 15;
  int b = wt >> 10;

  v8f acc = conv_c32_tile_bf(Mc, Wp, b, t, h0, w0, lane);

  float tl = softplusf(tau_l_p[0]);
  float rs = softplusf(rho_p[0]);
  float cu = tl / (rs + 1e-4f);
  float bias = b0[0];
  int n = lane & 15, mb = (lane >> 4) << 3;
#pragma unroll
  for (int jv = 0; jv < 8; ++jv) {
    int m = mb + jv;
    int idx = ((b * Tn + t) * Hn + h0 + m) * Wn + w0 + n;
    float bg = 1.f - (acc[jv] + bias);
    float bg2 = bg * bg;
    float xl = X[idx], ll = L[idx], uu = U[idx];
    Ltmp[idx] = ll - tl * bg2 * ll + tl * bg2 * xl - cu * bg * uu;
  }
}

// ---------------------------------------------------------------------------
// SVT stage: Gram matrix, Jacobi eigensolve, projection.
// ---------------------------------------------------------------------------
__global__ void k_gram(const float* __restrict__ Ltmp, float* __restrict__ G) {
  int b = blockIdx.x, tid = threadIdx.x;
  int i = tid >> 4, j = tid & 15;
  const float4* ri = (const float4*)(Ltmp + (b * Tn + i) * HWn);
  const float4* rj = (const float4*)(Ltmp + (b * Tn + j) * HWn);
  float acc = 0.f;
  for (int p = 0; p < HWn / 4; ++p) {
    float4 a = ri[p], c = rj[p];
    acc += a.x * c.x + a.y * c.y + a.z * c.z + a.w * c.w;
  }
  G[b * 256 + tid] = acc;
}

__global__ void k_eig(const float* __restrict__ G, const float* __restrict__ coefL_p,
                      float* __restrict__ P) {
  int b = threadIdx.x;
  if (b >= Bn) return;
  float A[16][16], V[16][16];
  for (int i = 0; i < 16; ++i)
    for (int j = 0; j < 16; ++j) {
      A[i][j] = G[b * 256 + i * 16 + j];
      V[i][j] = (i == j) ? 1.f : 0.f;
    }
  for (int sweep = 0; sweep < 12; ++sweep) {
    for (int p = 0; p < 15; ++p) {
      for (int q = p + 1; q < 16; ++q) {
        float apq = A[p][q];
        if (fabsf(apq) > 1e-12f) {
          float tau = (A[q][q] - A[p][p]) / (2.f * apq);
          float tt = ((tau >= 0.f) ? 1.f : -1.f) / (fabsf(tau) + sqrtf(1.f + tau * tau));
          float cc = 1.f / sqrtf(1.f + tt * tt);
          float ss = tt * cc;
          for (int k = 0; k < 16; ++k) {
            float akp = A[k][p], akq = A[k][q];
            A[k][p] = cc * akp - ss * akq;
            A[k][q] = ss * akp + cc * akq;
          }
          for (int k = 0; k < 16; ++k) {
            float apk = A[p][k], aqk = A[q][k];
            A[p][k] = cc * apk - ss * aqk;
            A[q][k] = ss * apk + cc * aqk;
          }
          for (int k = 0; k < 16; ++k) {
            float vkp = V[k][p], vkq = V[k][q];
            V[k][p] = cc * vkp - ss * vkq;
            V[k][q] = ss * vkp + cc * vkq;
          }
        }
      }
    }
  }
  float S[16], smax = 0.f;
  for (int i = 0; i < 16; ++i) {
    S[i] = sqrtf(fmaxf(A[i][i], 0.f));
    smax = fmaxf(smax, S[i]);
  }
  float thr = fmaxf(coefL_p[0], 0.f) * smax;
  float f[16];
  for (int i = 0; i < 16; ++i)
    f[i] = (S[i] > 1e-12f) ? fmaxf(S[i] - thr, 0.f) / S[i] : 0.f;
  for (int i = 0; i < 16; ++i)
    for (int j = 0; j < 16; ++j) {
      float acc = 0.f;
      for (int k = 0; k < 16; ++k) acc += V[i][k] * f[k] * V[j][k];
      P[b * 256 + i * 16 + j] = acc;
    }
}

__global__ void k_project(const float* __restrict__ P, const float* __restrict__ Ltmp,
                          const float* __restrict__ X, float* __restrict__ OutX,
                          float* __restrict__ OutL, float* __restrict__ fg) {
  int n = blockIdx.x * 256 + threadIdx.x;
  int p = n & 16383;
  int i = (n >> 14) & 15;
  int b = n >> 18;
  const float* Pb = P + b * 256 + i * 16;
  const float* Lb = Ltmp + b * Tn * HWn + p;
  float acc = 0.f;
#pragma unroll
  for (int j = 0; j < 16; ++j) acc += Pb[j] * Lb[j * HWn];
  float x = X[n];
  OutX[n] = x;
  OutL[n] = acc;
  fg[n] = x - acc;
}

// ---------------------------------------------------------------------------
// G1 = (1 - conv1(Mc)) * fg^2 ; G2 = fg * U  (padded bf16 outs with halos)
// ---------------------------------------------------------------------------
__global__ void k_conv1_g1(const unsigned short* __restrict__ Mc,
                           const float* __restrict__ W1, const float* __restrict__ fg,
                           const float* __restrict__ U, unsigned short* __restrict__ G1,
                           unsigned short* __restrict__ G2) {
  __shared__ unsigned int Wp[HIDn * 120];
  int tid = threadIdx.x;
  stage_wp5(Wp, W1, tid);
  __syncthreads();

  int lane = tid & 31;
  int wt = blockIdx.x * 8 + (tid >> 5);
  int w0 = (wt & 7) << 4;
  int h0 = ((wt >> 3) & 7) << 4;
  int t = (wt >> 6) & 15;
  int b = wt >> 10;

  v8f acc = conv_c32_tile_bf(Mc, Wp, b, t, h0, w0, lane);

  int n = lane & 15, mb = (lane >> 4) << 3;
#pragma unroll
  for (int jv = 0; jv < 8; ++jv) {
    int m = mb + jv;
    int idx = ((b * Tn + t) * Hn + h0 + m) * Wn + w0 + n;
    float f = fg[idx];
    unsigned short g1 = f2bf((1.f - acc[jv]) * f * f);
    unsigned short g2 = f2bf(f * U[idx]);
    size_t pi = (size_t)((b * Tn + t) * Hn + h0 + m) * RSp + 4 + w0 + n;
    G1[pi] = g1;
    G2[pi] = g2;
    if (w0 == 0 && n == 0) {
#pragma unroll
      for (int q = 0; q < 4; ++q) { G1[pi - 4 + q] = g1; G2[pi - 4 + q] = g2; }
    }
    if (w0 == 112 && n == 15) {
#pragma unroll
      for (int q = 0; q < 4; ++q) { G1[pi + 1 + q] = g1; G2[pi + 1 + q] = g2; }
    }
  }
}

// ---------------------------------------------------------------------------
// Mtmp = Mc + tau_s*conv2(G1) - (tau_s/(rho+eps))*conv3(G2)
// Two chained im2col GEMMs (K' layout) with pre-scaled pair-packed weights.
// ---------------------------------------------------------------------------
__global__ void k_conv23(const unsigned short* __restrict__ G1,
                         const unsigned short* __restrict__ G2,
                         const float* __restrict__ W2, const float* __restrict__ W3,
                         const unsigned short* __restrict__ Mc,
                         const float* __restrict__ tau_s_p, const float* __restrict__ rho_p,
                         float* __restrict__ Mtmp, unsigned short* __restrict__ MtmpBf) {
  __shared__ unsigned int B2s[64 * 32];
  __shared__ unsigned int B3s[64 * 32];
  float ts = softplusf(tau_s_p[0]);
  float rs = softplusf(rho_p[0]);
  float s3 = -ts / (rs + 1e-4f);
  int tid = threadIdx.x;
  for (int idx = tid; idx < 64 * 32; idx += 256) {
    int p = idx >> 5, o = idx & 31;
    B2s[idx] = pack2(wkp(W2, o, 2 * p, ts), wkp(W2, o, 2 * p + 1, ts));
    B3s[idx] = pack2(wkp(W3, o, 2 * p, s3), wkp(W3, o, 2 * p + 1, s3));
  }
  __syncthreads();

  int lane = tid & 31;
  int wt = blockIdx.x * 8 + (tid >> 5);
  int w0 = (wt & 7) << 4;
  int h = (wt >> 3) & 127;
  int t = (wt >> 10) & 15;
  int b = wt >> 14;
  int ar = lane & 15;
  int lo = (lane & 16) ? 8 : 0;
  int n0 = lane & 15;

  int rowbase[16];
  rowbase15(b, t, h, rowbase);
  int wcol = w0 + ar + 2;
  int pi8[8];
#pragma unroll
  for (int i = 0; i < 8; ++i) pi8[i] = pidx(lane, i);

  v8f acc0 = {}, acc1 = {};
#pragma unroll
  for (int kc = 0; kc < 4; ++kc) {
    int kb = kc * 32;
    int r1 = (kb + lo) >> 3;
    const unsigned short* p1a = G1 + rowbase[r1] + wcol;
    const unsigned short* p2a = G1 + rowbase[r1 + 2] + wcol;
    const unsigned short* p1b = G2 + rowbase[r1] + wcol;
    const unsigned short* p2b = G2 + rowbase[r1 + 2] + wcol;
    unsigned short au1[16], au2[16];
#pragma unroll
    for (int i = 0; i < 8; ++i) {
      au1[i] = p1a[i];
      au1[8 + i] = p2a[i];
      au2[i] = p1b[i];
      au2[8 + i] = p2b[i];
    }
    u32x8 a1v, a2v;
#pragma unroll
    for (int i = 0; i < 8; ++i) {
      a1v[i] = pack2(au1[2 * i], au1[2 * i + 1]);
      a2v[i] = pack2(au2[2 * i], au2[2 * i + 1]);
    }
    v16bf a1 = __builtin_bit_cast(v16bf, a1v);
    v16bf a2 = __builtin_bit_cast(v16bf, a2v);
    u32x8 bv;
#pragma unroll
    for (int i = 0; i < 8; ++i) bv[i] = B2s[(kc * 16 + pi8[i]) * 32 + n0];
    acc0 = wmma_bf16(a1, __builtin_bit_cast(v16bf, bv), acc0);
#pragma unroll
    for (int i = 0; i < 8; ++i) bv[i] = B3s[(kc * 16 + pi8[i]) * 32 + n0];
    acc0 = wmma_bf16(a2, __builtin_bit_cast(v16bf, bv), acc0);
#pragma unroll
    for (int i = 0; i < 8; ++i) bv[i] = B2s[(kc * 16 + pi8[i]) * 32 + n0 + 16];
    acc1 = wmma_bf16(a1, __builtin_bit_cast(v16bf, bv), acc1);
#pragma unroll
    for (int i = 0; i < 8; ++i) bv[i] = B3s[(kc * 16 + pi8[i]) * 32 + n0 + 16];
    acc1 = wmma_bf16(a2, __builtin_bit_cast(v16bf, bv), acc1);
  }

  int mb = (lane >> 4) << 3;
  size_t pb0 = (size_t)(b * HIDn + n0) * CSp + (size_t)t * PLp + (size_t)h * RSp + 4 + w0;
  size_t pb1 = pb0 + (size_t)16 * CSp;
  int u0 = (((b * HIDn + n0) * Tn + t) * Hn + h) * Wn + w0;
  int u1 = (((b * HIDn + n0 + 16) * Tn + t) * Hn + h) * Wn + w0;
  float v0l = 0.f, v1l = 0.f, v0r = 0.f, v1r = 0.f;
#pragma unroll
  for (int jv = 0; jv < 8; ++jv) {
    int mo = mb + jv;
    float v0 = bf2f(Mc[pb0 + mo]) + acc0[jv];
    float v1 = bf2f(Mc[pb1 + mo]) + acc1[jv];
    Mtmp[u0 + mo] = v0;
    Mtmp[u1 + mo] = v1;
    MtmpBf[pb0 + mo] = f2bf(v0);
    MtmpBf[pb1 + mo] = f2bf(v1);
    if (jv == 0) { v0l = v0; v1l = v1; }
    if (jv == 7) { v0r = v0; v1r = v1; }
  }
  if (w0 == 0 && mb == 0) {
    unsigned short e0 = f2bf(v0l), e1 = f2bf(v1l);
#pragma unroll
    for (int q = 0; q < 4; ++q) { MtmpBf[pb0 - 4 + q] = e0; MtmpBf[pb1 - 4 + q] = e1; }
  }
  if (w0 == 112 && mb == 8) {
    unsigned short e0 = f2bf(v0r), e1 = f2bf(v1r);
#pragma unroll
    for (int q = 0; q < 4; ++q) { MtmpBf[pb0 + 16 + q] = e0; MtmpBf[pb1 + 16 + q] = e1; }
  }
}

// ---------------------------------------------------------------------------
// Depthwise 7x7 conv of t-1 shifted Mtmp (Toeplitz WMMA) fused with soft_l1_l1.
// Pair-packed weights: [c][dh][10 words], pair[d'] = (w[d'-1], w[d']); 8 = zero.
// Output (padded bf16, halos) reuses the dead Mc buffer.
// ---------------------------------------------------------------------------
__global__ void k_prox(const float* __restrict__ Mtmp,
                       const unsigned short* __restrict__ MtmpBf,
                       const float* __restrict__ WG, const float* __restrict__ cS_p,
                       const float* __restrict__ cSS_p, unsigned short* __restrict__ Sout) {
  __shared__ unsigned int Wp[HIDn * 70];  // 7 dh * 10 words
  int tid = threadIdx.x;
  for (int idx = tid; idx < HIDn * 70; idx += 256) {
    int c = idx / 70, r = idx - c * 70;
    int dh = r / 10, dp = r - dh * 10;
    const float* wr = WG + c * 49 + dh * 7;
    unsigned short lo = (dp >= 1 && dp <= 7) ? f2bf(wr[dp - 1]) : (unsigned short)0;
    unsigned short hi = (dp < 7) ? f2bf(wr[dp]) : (unsigned short)0;
    Wp[idx] = pack2(lo, hi);
  }
  __syncthreads();

  int lane = tid & 31;
  int wt = blockIdx.x * 8 + (tid >> 5);
  int w0 = (wt & 7) << 4;
  int h0 = ((wt >> 3) & 7) << 4;
  int t = (wt >> 6) & 15;
  int c = (wt >> 10) & 31;
  int b = wt >> 15;
  int ts_ = (t == 0) ? 0 : t - 1;

  int ar = lane & 15, j = lane & 15;
  int woff[8], psel[8];
#pragma unroll
  for (int i = 0; i < 8; ++i) {
    int p = pidx(lane, i);
    woff[i] = w0 + 2 * p + 1;  // padded col of pair low half (kk=2p): 4+w0+2p-3
    int d = 2 * p - j;
    psel[i] = (d >= -1 && d <= 6) ? (d + 1) : 8;  // 8 -> zero word
  }
  int hrow[7];
#pragma unroll
  for (int dh = 0; dh < 7; ++dh) hrow[dh] = iclamp(h0 + ar + dh - 3, 0, Hn - 1) * RSp;

  const unsigned short* src = MtmpBf + ((size_t)(b * HIDn + c) * Tn + ts_) * PLp;
  const unsigned int* wc = Wp + c * 70;
  v8f acc = {};
  for (int dh = 0; dh < 7; ++dh) {
    const unsigned short* row = src + hrow[dh];
    const unsigned int* wrow = wc + dh * 10;
    u32x8 av, bv;
#pragma unroll
    for (int i = 0; i < 8; ++i) {
      av[i] = pack2(row[woff[i]], row[woff[i] + 1]);
      bv[i] = wrow[psel[i]];
    }
    acc = wmma_bf16(__builtin_bit_cast(v16bf, av), __builtin_bit_cast(v16bf, bv), acc);
  }

  float w0c = cS_p[0], w1c = cSS_p[0];
  const float* zsrc = Mtmp + ((size_t)(b * HIDn + c) * Tn + t) * HWn;
  unsigned short* dst = Sout + ((size_t)(b * HIDn + c) * Tn + t) * PLp;
  int n = lane & 15, mb = (lane >> 4) << 3;
#pragma unroll
  for (int jv = 0; jv < 8; ++jv) {
    int m = mb + jv;
    float z = zsrc[(h0 + m) * Wn + w0 + n];
    float a1v = acc[jv];
    bool cond = a1v >= 0.f;
    float a0 = cond ? 0.f : a1v;
    float a1 = cond ? a1v : 0.f;
    float w0s = cond ? w0c : w1c;
    float w1s = cond ? w1c : w0c;
    float r;
    if (z >= a1 + w0s + w1s) r = z - w0s - w1s;
    else if (z >= a1 + w0s - w1s) r = a1;
    else if (z >= a0 + w0s - w1s) r = z - w0s + w1s;
    else if (z >= a0 - w0s - w1s) r = a0;
    else r = z + w0s + w1s;
    unsigned short rb = f2bf(r);
    size_t rowp = (size_t)(h0 + m) * RSp;
    dst[rowp + 4 + w0 + n] = rb;
    if (w0 == 0 && n == 0) {
#pragma unroll
      for (int q = 0; q < 4; ++q) dst[rowp + q] = rb;
    }
    if (w0 == 112 && n == 15) {
#pragma unroll
      for (int q = 0; q < 4; ++q) dst[rowp + 132 + q] = rb;
    }
  }
}

// ---------------------------------------------------------------------------
// Mout = sigmoid((conv4(S)-0.5)*ms); Unew = U + rho*(1-Mout)*(Lnew-X)  (= -fg)
// ---------------------------------------------------------------------------
__global__ void k_conv4_out(const unsigned short* __restrict__ S,
                            const float* __restrict__ W4, const float* __restrict__ U,
                            const float* __restrict__ fg, const float* __restrict__ ms_p,
                            const float* __restrict__ rho_p, float* __restrict__ Mout,
                            float* __restrict__ Unew) {
  __shared__ unsigned int Wp[HIDn * 120];
  int tid = threadIdx.x;
  stage_wp5(Wp, W4, tid);
  __syncthreads();

  int lane = tid & 31;
  int wt = blockIdx.x * 8 + (tid >> 5);
  int w0 = (wt & 7) << 4;
  int h0 = ((wt >> 3) & 7) << 4;
  int t = (wt >> 6) & 15;
  int b = wt >> 10;

  v8f acc = conv_c32_tile_bf(S, Wp, b, t, h0, w0, lane);

  float ms = ms_p[0];
  float rs = softplusf(rho_p[0]);
  int n = lane & 15, mb = (lane >> 4) << 3;
#pragma unroll
  for (int jv = 0; jv < 8; ++jv) {
    int m = mb + jv;
    int idx = ((b * Tn + t) * Hn + h0 + m) * Wn + w0 + n;
    float mo = 1.f / (1.f + expf(-(acc[jv] - 0.5f) * ms));
    Mout[idx] = mo;
    Unew[idx] = U[idx] + rs * (1.f - mo) * (-fg[idx]);
  }
}

// ---------------------------------------------------------------------------
extern "C" void kernel_launch(void* const* d_in, const int* in_sizes, int n_in,
                              void* d_out, int out_size, void* d_ws, size_t ws_size,
                              hipStream_t stream) {
  (void)in_sizes; (void)n_in; (void)out_size; (void)ws_size;
  const float* X = (const float*)d_in[0];
  const float* L = (const float*)d_in[1];
  const float* M = (const float*)d_in[2];
  const float* U = (const float*)d_in[3];
  const float* WconvM = (const float*)d_in[4];
  const float* Wconv0 = (const float*)d_in[5];
  const float* bconv0 = (const float*)d_in[6];
  const float* Wconv1 = (const float*)d_in[7];
  const float* Wconv2 = (const float*)d_in[8];
  const float* Wconv3 = (const float*)d_in[9];
  const float* Wconv4 = (const float*)d_in[10];
  const float* WG = (const float*)d_in[11];
  const float* tau_l = (const float*)d_in[12];
  const float* tau_s = (const float*)d_in[13];
  const float* rho = (const float*)d_in[14];
  const float* msc = (const float*)d_in[15];
  const float* coefL = (const float*)d_in[16];
  const float* coefS = (const float*)d_in[17];
  const float* coefSS = (const float*)d_in[18];

  float* out = (float*)d_out;  // [X | Lnew | Mout | Unew], each NPIX
  char* wsb = (char*)d_ws;
  unsigned short* McP = (unsigned short*)wsb;                 // padded 32-ch bf16
  unsigned short* MtmpBfP = McP + PADBUF_HALFS;               // padded 32-ch bf16
  float* Mtmp = (float*)(wsb + (size_t)PADBUF_HALFS * 4);     // NCPIX f32
  float* Ltmp = Mtmp + (size_t)NCPIX;                         // NPIX f32
  float* fg = Ltmp + NPIX;                                    // NPIX f32
  unsigned short* Mp = (unsigned short*)(fg + NPIX);          // padded 1-ch bf16
  unsigned short* G1p = Mp + SPADBUF_HALFS;                   // padded 1-ch bf16
  unsigned short* G2p = G1p + SPADBUF_HALFS;                  // padded 1-ch bf16
  float* gram = (float*)(G2p + SPADBUF_HALFS);                // 4*256
  float* P = gram + 1024;                                     // 4*256

  k_slack<<<1, 128, 0, stream>>>(McP, MtmpBfP, Mp, G1p, G2p);
  k_tobf16_pad<<<NPIX / 256, 256, 0, stream>>>(M, Mp);
  k_convM<<<8192, 256, 0, stream>>>(Mp, WconvM, McP);
  k_conv0_ltmp<<<512, 256, 0, stream>>>(McP, Wconv0, bconv0, X, L, U, tau_l, rho, Ltmp);
  k_gram<<<4, 256, 0, stream>>>(Ltmp, gram);
  k_eig<<<1, 32, 0, stream>>>(gram, coefL, P);
  k_project<<<4096, 256, 0, stream>>>(P, Ltmp, X, out, out + NPIX, fg);
  k_conv1_g1<<<512, 256, 0, stream>>>(McP, Wconv1, fg, U, G1p, G2p);
  k_conv23<<<8192, 256, 0, stream>>>(G1p, G2p, Wconv2, Wconv3, McP, tau_s, rho, Mtmp,
                                     MtmpBfP);
  k_prox<<<16384, 256, 0, stream>>>(Mtmp, MtmpBfP, WG, coefS, coefSS, McP);
  k_conv4_out<<<512, 256, 0, stream>>>(McP, Wconv4, U, fg, msc, rho,
                                       out + 2 * (size_t)NPIX, out + 3 * (size_t)NPIX);
}
